// GCN_8340826489039
// MI455X (gfx1250) — matrix operand order repaired and checked by reading, not compiled, not run on hardware
//
#include <hip/hip_runtime.h>
#include <hip/hip_bf16.h>

// ---------------------------------------------------------------------------
// Types for CDNA5 WMMA
// ---------------------------------------------------------------------------
typedef __attribute__((ext_vector_type(16))) __bf16 v16bf;
typedef __attribute__((ext_vector_type(8)))  float  v8f;

// Pack two floats into one DWORD of bf16 using native converts (v_cvt_pk_bf16_f32).
__device__ __forceinline__ unsigned gcn_pack_bf2(float a, float b) {
    union { __bf16 h[2]; unsigned u; } cv;
    cv.h[0] = (__bf16)a;
    cv.h[1] = (__bf16)b;
    return cv.u;
}

#define GCN_MTILE 128
#define GCN_KTILE 32

// ---------------------------------------------------------------------------
// Weight pack kernel: W[K,Nout] fp32 -> PW packed bf16 pairs in exact WMMA
// B-fragment order:  dword index = ((s*NT + t)*32 + lane)*8 + v
//   n  = t*16 + (lane&15),  k0 = s*32 + (lane>>4)*16 + v*2   (zero padded)
// ---------------------------------------------------------------------------
__global__ __launch_bounds__(256) void gcn_pack_w(
    const float* __restrict__ W, unsigned* __restrict__ PW,
    int K, int Nout, int NT, int Ksteps)
{
    int gid = blockIdx.x * 256 + threadIdx.x;
    int total = Ksteps * NT * 256;
    if (gid >= total) return;
    int v    = gid & 7;
    int lane = (gid >> 3) & 31;
    int st   = gid >> 8;            // s*NT + t
    int t    = st % NT;
    int s    = st / NT;
    int n    = t * 16 + (lane & 15);
    int k0   = s * 32 + (lane >> 4) * 16 + v * 2;
    float f0 = 0.0f, f1 = 0.0f;
    if (n < Nout) {
        if (k0     < K) f0 = W[(size_t)k0 * Nout + n];
        if (k0 + 1 < K) f1 = W[(size_t)(k0 + 1) * Nout + n];
    }
    PW[gid] = gcn_pack_bf2(f0, f1);
}

// ---------------------------------------------------------------------------
// WMMA GEMM:  C[M,Nout] = A[M,K] @ W[K,Nout]  (fp32 A from HBM, bf16 WMMA)
// 256 threads = 8 waves; block tile 128 rows x NTILES*16 cols.
// Software-pipelined on two levels:
//   - A tile s+1 global loads in flight during tile s WMMAs (double-buffered
//     LDS, 2 barriers / K-step)
//   - B fragments double-buffered in registers so load(t+1) overlaps wmma(t)
//     instead of forming a WAR chain on a single register set.
// ---------------------------------------------------------------------------
template <int NTILES>
__global__ __launch_bounds__(256) void gcn_gemm_wmma_bf16(
    const float* __restrict__ A, const unsigned* __restrict__ PW,
    float* __restrict__ C, int M, int K, int Nout)
{
    __shared__ unsigned sA[2][GCN_MTILE * 17];   // [buf][row*17 + pair]

    const int tid    = threadIdx.x;
    const int lane   = tid & 31;
    const int wave   = tid >> 5;      // 0..7
    const int half   = lane >> 4;     // 0/1
    const int mlane  = lane & 15;
    const int rowBlk = blockIdx.x * GCN_MTILE;

    v8f acc[NTILES];
#pragma unroll
    for (int t = 0; t < NTILES; ++t)
#pragma unroll
        for (int j = 0; j < 8; ++j) acc[t][j] = 0.0f;

    union FragA { unsigned u[8]; v16bf v; };
    union FragB { uint4 q[2];   v16bf v; };
    const uint4* pwQ = reinterpret_cast<const uint4*>(PW);

    // Per-thread staging coords: fixed pair p, rows r0 + 16*i (row-clamped;
    // out-of-range rows compute garbage that is masked at store time).
    const int p  = tid & 15;
    const int r0 = tid >> 4;
    const float* Aptr[8];
#pragma unroll
    for (int i = 0; i < 8; ++i) {
        int grow = rowBlk + r0 + i * 16;
        grow = grow < M ? grow : M - 1;
        Aptr[i] = A + (size_t)grow * K + p * 2;
    }

    const int kFull = K & ~(GCN_KTILE - 1);
    const int steps = kFull >> 5;

    float2 G[8];
    if (steps > 0) {
#pragma unroll
        for (int i = 0; i < 8; ++i)
            G[i] = *reinterpret_cast<const float2*>(Aptr[i]);
    }

    for (int s = 0; s < steps; ++s) {
        unsigned* buf = sA[s & 1];
        // ---- convert + store tile s
#pragma unroll
        for (int i = 0; i < 8; ++i)
            buf[(r0 + i * 16) * 17 + p] = gcn_pack_bf2(G[i].x, G[i].y);

        // ---- issue A loads for tile s+1 (clamped redundant reload on last)
        const int knext = (s + 1 < steps ? s + 1 : s) * GCN_KTILE;
#pragma unroll
        for (int i = 0; i < 8; ++i)
            G[i] = *reinterpret_cast<const float2*>(Aptr[i] + knext);

        __syncthreads();

        // ---- first B fragment of this step (overlaps A-fragment ds reads)
        FragB fb[2];
        {
            size_t qidx = ((size_t)(s * NTILES) * 32 + lane) * 2;
            fb[0].q[0] = pwQ[qidx];
            fb[0].q[1] = pwQ[qidx + 1];
        }

        // ---- A fragment (contiguous k = 16*half + element, same order as PW)
        FragA fa;
        const int arow = wave * 16 + mlane;
#pragma unroll
        for (int v = 0; v < 8; ++v)
            fa.u[v] = buf[arow * 17 + (half * 8 + v)];

        // ---- WMMAs with register-double-buffered B fragments
#pragma unroll
        for (int t = 0; t < NTILES; ++t) {
            if (t + 1 < NTILES) {
                size_t qidx = ((size_t)(s * NTILES + t + 1) * 32 + lane) * 2;
                fb[(t + 1) & 1].q[0] = pwQ[qidx];
                fb[(t + 1) & 1].q[1] = pwQ[qidx + 1];
            }
            acc[t] = __builtin_amdgcn_wmma_f32_16x16x32_bf16(
                false, fa.v, false, fb[t & 1].v, (short)0, acc[t], false, false);
        }
        // second barrier per step: separates this step's fragment reads of
        // buf[s&1] from step s+2's stores into the same buffer
        __syncthreads();
    }

    // ---- guarded tail (K % 32 != 0); PW step `steps` is zero-padded
    if (kFull < K) {
        unsigned* buf = sA[steps & 1];
#pragma unroll
        for (int i = 0; i < 8; ++i) {
            int gk = kFull + p * 2;
            float f0 = (gk     < K) ? Aptr[i][kFull + 0] : 0.0f;
            float f1 = (gk + 1 < K) ? Aptr[i][kFull + 1] : 0.0f;
            buf[(r0 + i * 16) * 17 + p] = gcn_pack_bf2(f0, f1);
        }
        __syncthreads();

        FragA fa;
        const int arow = wave * 16 + mlane;
#pragma unroll
        for (int v = 0; v < 8; ++v)
            fa.u[v] = buf[arow * 17 + (half * 8 + v)];
#pragma unroll
        for (int t = 0; t < NTILES; ++t) {
            FragB fb;
            size_t qidx = ((size_t)(steps * NTILES + t) * 32 + lane) * 2;
            fb.q[0] = pwQ[qidx];
            fb.q[1] = pwQ[qidx + 1];
            acc[t] = __builtin_amdgcn_wmma_f32_16x16x32_bf16(
                false, fa.v, false, fb.v, (short)0, acc[t], false, false);
        }
    }

    // ---- store (C/D layout: VGPR v -> row v + half*8, lane -> col)
#pragma unroll
    for (int t = 0; t < NTILES; ++t) {
        int n = t * 16 + mlane;
        if (n < Nout) {
#pragma unroll
            for (int v = 0; v < 8; ++v) {
                int grow = rowBlk + wave * 16 + v + half * 8;
                if (grow < M) C[(size_t)grow * Nout + n] = acc[t][v];
            }
        }
    }
}

// ---------------------------------------------------------------------------
// Scatter-add:  agg[dst[e], f] += h[src[e], f] * w[e]
// One thread per (edge, feature-pair); D compile-time -> div/mod are cheap.
// h and agg are L2-resident (N*D fp32 << 192 MB).
// ---------------------------------------------------------------------------
template <int D>
__global__ __launch_bounds__(256) void gcn_scatter_add2(
    const float* __restrict__ h, const int* __restrict__ src,
    const int* __restrict__ dst, const float* __restrict__ w,
    float* __restrict__ agg, unsigned total2)
{
    unsigned gid = blockIdx.x * 256u + threadIdx.x;
    if (gid >= total2) return;
    constexpr unsigned HD = (unsigned)D / 2u;
    unsigned e = gid / HD;
    unsigned f = (gid - e * HD) * 2u;
    float2 v = *reinterpret_cast<const float2*>(h + (size_t)src[e] * D + f);
    float ww = w[e];
    float* a = agg + (size_t)dst[e] * D + f;
    atomicAdd(a,     v.x * ww);
    atomicAdd(a + 1, v.y * ww);
}

__global__ __launch_bounds__(256) void gcn_scatter_add_gen(
    const float* __restrict__ h, const int* __restrict__ src,
    const int* __restrict__ dst, const float* __restrict__ w,
    float* __restrict__ agg, unsigned total, int D)
{
    unsigned gid = blockIdx.x * 256u + threadIdx.x;
    if (gid >= total) return;
    unsigned e = gid / (unsigned)D;
    unsigned f = gid - e * (unsigned)D;
    float val = h[(size_t)src[e] * D + f] * w[e];
    atomicAdd(&agg[(size_t)dst[e] * D + f], val);
}

// ---------------------------------------------------------------------------
// Elementwise helpers
// ---------------------------------------------------------------------------
__global__ __launch_bounds__(256) void gcn_fill_zero(float* __restrict__ p, unsigned n)
{
    unsigned gid = blockIdx.x * 256u + threadIdx.x;
    if (gid < n) p[gid] = 0.0f;
}

__global__ __launch_bounds__(256) void gcn_bias_act(
    float* __restrict__ y, const float* __restrict__ b, unsigned n, int D, int relu)
{
    unsigned gid = blockIdx.x * 256u + threadIdx.x;
    if (gid >= n) return;
    float v = y[gid] + b[gid % (unsigned)D];
    if (relu) v = v > 0.0f ? v : 0.0f;
    y[gid] = v;
}

// ---------------------------------------------------------------------------
// Host-side dispatch
// ---------------------------------------------------------------------------
static inline unsigned cdiv(unsigned a, unsigned b) { return (a + b - 1) / b; }

static int pick_nt(int Nout) {
    int nt = (Nout + 15) / 16;
    if (nt == 4 || nt == 5 || nt == 8) return nt;
    return 8;   // covers any Nout <= 128 (stores are masked)
}

static void launch_gemm(const float* A, const unsigned* PW, float* C,
                        int M, int K, int Nout, int nt, hipStream_t s)
{
    dim3 grid((M + GCN_MTILE - 1) / GCN_MTILE), block(256);
    switch (nt) {
        case 4: gcn_gemm_wmma_bf16<4><<<grid, block, 0, s>>>(A, PW, C, M, K, Nout); break;
        case 5: gcn_gemm_wmma_bf16<5><<<grid, block, 0, s>>>(A, PW, C, M, K, Nout); break;
        default: gcn_gemm_wmma_bf16<8><<<grid, block, 0, s>>>(A, PW, C, M, K, Nout); break;
    }
}

static void launch_scatter(const float* h, const int* src, const int* dst,
                           const float* w, float* agg, int Etot, int D,
                           hipStream_t s)
{
    dim3 block(256);
    if (D == 128 || D == 64 || D == 70) {
        unsigned total2 = (unsigned)Etot * (unsigned)(D / 2);
        dim3 grid(cdiv(total2, 256));
        switch (D) {
            case 128: gcn_scatter_add2<128><<<grid, block, 0, s>>>(h, src, dst, w, agg, total2); break;
            case 64:  gcn_scatter_add2<64> <<<grid, block, 0, s>>>(h, src, dst, w, agg, total2); break;
            case 70:  gcn_scatter_add2<70> <<<grid, block, 0, s>>>(h, src, dst, w, agg, total2); break;
        }
    } else {
        unsigned total = (unsigned)Etot * (unsigned)D;
        dim3 grid(cdiv(total, 256));
        gcn_scatter_add_gen<<<grid, block, 0, s>>>(h, src, dst, w, agg, total, D);
    }
}

extern "C" void kernel_launch(void* const* d_in, const int* in_sizes, int n_in,
                              void* d_out, int out_size, void* d_ws, size_t ws_size,
                              hipStream_t stream)
{
    const float* x   = (const float*)d_in[0];
    const int*   src = (const int*)  d_in[1];
    const int*   dst = (const int*)  d_in[2];
    const float* ew  = (const float*)d_in[3];
    const float* W1  = (const float*)d_in[4];
    const float* b1  = (const float*)d_in[5];
    const float* W2  = (const float*)d_in[6];
    const float* b2  = (const float*)d_in[7];
    const float* W3  = (const float*)d_in[8];
    const float* b3  = (const float*)d_in[9];
    float* out = (float*)d_out;

    const int H1   = in_sizes[5];              // 128
    const int F_IN = in_sizes[4] / H1;         // 8710
    const int Nn   = in_sizes[0] / F_IN;       // 50000
    const int Etot = in_sizes[1];              // E + N
    const int H2   = in_sizes[7];              // 64
    const int Cc   = in_sizes[9];              // 70

    const int nt1 = pick_nt(H1), nt2 = pick_nt(H2), nt3 = pick_nt(Cc);
    const int ks1 = (F_IN + 31) / 32, ks2 = (H1 + 31) / 32, ks3 = (H2 + 31) / 32;

    // ---- workspace layout
    float* B0 = (float*)d_ws;                              // Nn*H1 fp32
    float* B1 = B0 + (size_t)Nn * H1;                      // Nn*H1 fp32
    unsigned* PW1 = (unsigned*)(B1 + (size_t)Nn * H1);     // ks1*nt1*256 dwords
    unsigned* PW2 = PW1 + (size_t)ks1 * nt1 * 256;         // ks2*nt2*256
    unsigned* PW3 = PW2 + (size_t)ks2 * nt2 * 256;         // ks3*nt3*256

    // ---- pack all weights (tiny, L2-resident afterwards)
    gcn_pack_w<<<cdiv((unsigned)ks1 * nt1 * 256, 256), 256, 0, stream>>>(W1, PW1, F_IN, H1, nt1, ks1);
    gcn_pack_w<<<cdiv((unsigned)ks2 * nt2 * 256, 256), 256, 0, stream>>>(W2, PW2, H1,   H2, nt2, ks2);
    gcn_pack_w<<<cdiv((unsigned)ks3 * nt3 * 256, 256), 256, 0, stream>>>(W3, PW3, H2,   Cc, nt3, ks3);

    // ---------------- Layer 1 ----------------
    unsigned n1 = (unsigned)Nn * H1;
    gcn_fill_zero<<<cdiv(n1, 256), 256, 0, stream>>>(B1, n1);
    launch_gemm(x, PW1, B0, Nn, F_IN, H1, nt1, stream);              // h1 = x @ W1
    launch_scatter(B0, src, dst, ew, B1, Etot, H1, stream);
    gcn_bias_act<<<cdiv(n1, 256), 256, 0, stream>>>(B1, b1, n1, H1, 1);

    // ---------------- Layer 2 ----------------
    launch_gemm(B1, PW2, B0, Nn, H1, H2, nt2, stream);               // h2 = h1r @ W2
    float* agg2 = B0 + (size_t)Nn * H2;
    unsigned n2 = (unsigned)Nn * H2;
    gcn_fill_zero<<<cdiv(n2, 256), 256, 0, stream>>>(agg2, n2);
    launch_scatter(B0, src, dst, ew, agg2, Etot, H2, stream);
    gcn_bias_act<<<cdiv(n2, 256), 256, 0, stream>>>(agg2, b2, n2, H2, 1);

    // ---------------- Layer 3 ----------------
    launch_gemm(agg2, PW3, B1, Nn, H2, Cc, nt3, stream);             // h3 = h2r @ W3
    unsigned n3 = (unsigned)Nn * Cc;
    gcn_fill_zero<<<cdiv(n3, 256), 256, 0, stream>>>(out, n3);
    launch_scatter(B1, src, dst, ew, out, Etot, Cc, stream);
    gcn_bias_act<<<cdiv(n3, 256), 256, 0, stream>>>(out, b3, n3, Cc, 0);
}
